// Block_4690104287929
// MI455X (gfx1250) — matrix operand order repaired and checked by reading, not compiled, not run on hardware
//
#include <hip/hip_runtime.h>
#include <hip/hip_bf16.h>

typedef _Float16 h16;
typedef __attribute__((ext_vector_type(4)))  _Float16 v4h;
typedef __attribute__((ext_vector_type(8)))  _Float16 v8h;
typedef __attribute__((ext_vector_type(16))) _Float16 v16h;
typedef __attribute__((ext_vector_type(8)))  float    v8f;

#define DEVI static __device__ __forceinline__

constexpr int B_  = 4;
constexpr int S_  = 2048;
constexpr int H_  = 1024;
constexpr int NH_ = 16;
constexpr int HD_ = 64;
constexpr int FF_ = 4096;
constexpr int M_  = B_ * S_;   // 8192 rows

// ---------------- CDNA5 feature detection ----------------
#if defined(__has_builtin)
#  if __has_builtin(__builtin_amdgcn_tensor_load_to_lds)
#    define HAVE_TDM 1
#  else
#    define HAVE_TDM 0
#  endif
#  if __has_builtin(__builtin_amdgcn_s_wait_tensorcnt)
#    define WAIT_TENSORCNT0() __builtin_amdgcn_s_wait_tensorcnt(0)
#  else
#    define WAIT_TENSORCNT0() asm volatile("s_wait_tensorcnt 0" ::: "memory")
#  endif
#else
#  define HAVE_TDM 0
#  define WAIT_TENSORCNT0() asm volatile("s_wait_tensorcnt 0" ::: "memory")
#endif

DEVI v16h cat8(v8h lo, v8h hi) {
  return __builtin_shufflevector(lo, hi, 0,1,2,3,4,5,6,7,8,9,10,11,12,13,14,15);
}

DEVI v8f wmma_f16(v16h a, v16h b, v8f c) {
  // D = A(16x32 f16) * B(32x16 f16) + C(16x16 f32)
  return __builtin_amdgcn_wmma_f32_16x16x32_f16(
      /*neg_a=*/false, a, /*neg_b=*/false, b,
      /*c_mod=*/(short)0, c, /*reuse_a=*/false, /*reuse_b=*/false);
}

// tanh-approx GELU with cheap exp-based tanh (clamped; exact in-range)
DEVI float gelu_tanh_f(float x) {
  float t = 0.7978845608028654f * (x + 0.044715f * x * x * x);
  t = fminf(fmaxf(t, -9.0f), 9.0f);
  float e = __expf(2.0f * t);
  float th = (e - 1.0f) / (e + 1.0f);
  return 0.5f * x * (1.0f + th);
}

#if HAVE_TDM
typedef __attribute__((ext_vector_type(4))) unsigned int u32x4;
typedef __attribute__((ext_vector_type(8))) int          i32x8;
typedef __attribute__((ext_vector_type(4))) int          i32x4;

// TDM 2D tile load: tile = 32 (contiguous k) x 128 rows of f16, row stride = Kdim,
// with LDS padding 4 DWORDs every 16 DWORDs  ->  LDS rows of 40 halves (80 B).
// D# field packing per CDNA5 ISA ch.8 (group0 128b, group1 256b; groups 2/3 zero => 2D).
DEVI void tdm_load_tile_2d(const h16* gsrc, unsigned lds_off, int Kdim) {
  unsigned long long ga = (unsigned long long)(uintptr_t)gsrc;
  u32x4 g0;
  g0[0] = 1u;                                              // count=1, user-mode, no gather
  g0[1] = lds_off;                                         // lds_addr [63:32]
  g0[2] = (unsigned)ga;                                    // global_addr [95:64]
  g0[3] = (unsigned)((ga >> 32) & 0x01ffffffull)           // global_addr [120:96]
        | (2u << 30);                                      // type=2 ("image") [127:126]
  unsigned td0 = (unsigned)Kdim;                           // tensor_dim0
  unsigned td1 = 1u << 20;                                 // tensor_dim1 (ample)
  unsigned long long st0 = (unsigned long long)(unsigned)Kdim;  // dim0 stride (elems)
  i32x8 g1;
  g1[0] = (int)((1u << 16)    // data_size = 1 (2 bytes)
              | (1u << 20)    // pad_enable
              | (3u << 22)    // pad_interval: 16 DWORDs
              | (3u << 25));  // pad_amount: 4 DWORDs
  g1[1] = (int)((td0 & 0xffffu) << 16);                    // tensor_dim0 lo16 @ [63:48]
  g1[2] = (int)((td0 >> 16) | ((td1 & 0xffffu) << 16));    // dim0 hi16, dim1 lo16
  g1[3] = (int)((td1 >> 16) | (32u << 16));                // dim1 hi16, tile_dim0=32
  g1[4] = (int)(128u);                                     // tile_dim1=128, tile_dim2=0
  g1[5] = (int)(st0 & 0xffffffffull);                      // dim0_stride lo32
  g1[6] = (int)((st0 >> 32) & 0xffffull);                  // dim0_stride hi16, dim1_stride lo=0
  g1[7] = 0;
  i32x4 z4 = (i32x4)0;
#if __clang_major__ >= 23
  i32x8 z8 = (i32x8)0;
  __builtin_amdgcn_tensor_load_to_lds(g0, g1, z4, z4, z8, 0);
#else
  __builtin_amdgcn_tensor_load_to_lds(g0, g1, z4, z4, 0);
#endif
}
#endif  // HAVE_TDM

// ---------------------------------------------------------------------------
// fp32 -> fp16 converter (vectorized: float4 in, 4 halves out)
// ---------------------------------------------------------------------------
__global__ void cvt4_kernel(const float4* __restrict__ in, h16* __restrict__ out, int n4) {
  int i = blockIdx.x * blockDim.x + threadIdx.x;
  if (i >= n4) return;
  float4 v = in[i];
  v4h h;
  h[0] = (h16)v.x; h[1] = (h16)v.y; h[2] = (h16)v.z; h[3] = (h16)v.w;
  *(v4h*)(out + (size_t)i * 4) = h;
}

// ---------------------------------------------------------------------------
// Tiled WMMA GEMM:  C[M,N] = A[M,K] @ B[K,N] + bias   (A,B f16 row-major)
// Block 256 threads = 8 waves, block tile 128x128, K-step 32, double-buffered.
// A tiles: TDM async DMA (wave 0 issues, TENSORcnt), LDS padding done by TDM.
// B tiles: register-staged (global loads issued early, transposed LDS store late)
//          so both streams overlap the WMMA work of the current buffers.
// ---------------------------------------------------------------------------
template<bool OUT_HALF, bool GELU>
__global__ __launch_bounds__(256)
void gemm_f16(const h16* __restrict__ A, const h16* __restrict__ Bm,
              const float* __restrict__ bias, void* __restrict__ outp,
              int M, int N, int K, float scale)
{
  __shared__ h16 As[2][128][40];   // [m][k], 80B row stride (b128-aligned frags)
  __shared__ h16 Bt[2][128][40];   // [n][k] transposed

  const int tid  = threadIdx.x;
  const int lane = tid & 31;
  const int wid  = tid >> 5;
  const int l16  = lane & 15;
  const int hf   = lane >> 4;
  const int wm   = wid >> 1;       // 0..3
  const int wn   = wid & 1;        // 0..1
  const int row0 = blockIdx.y * 128;
  const int col0 = blockIdx.x * 128;

  v8h breg[2];

  auto stage_A = [&](int buf, int k0) {
#if HAVE_TDM
    if (wid == 0)
      tdm_load_tile_2d(A + (size_t)row0 * K + k0,
                       (unsigned)(uintptr_t)&As[buf][0][0], K);
#else
    #pragma unroll
    for (int i = 0; i < 2; ++i) {
      int idx = tid + i * 256;
      int r   = idx >> 2;           // 0..127
      int c   = (idx & 3) * 8;      // 0,8,16,24
      *(v8h*)&As[buf][r][c] = *(const v8h*)(A + (size_t)(row0 + r) * K + k0 + c);
    }
#endif
  };
  auto wait_A = [&]() {
#if HAVE_TDM
    if (wid == 0) WAIT_TENSORCNT0();
#endif
  };
  auto load_B = [&](int k0) {
    #pragma unroll
    for (int i = 0; i < 2; ++i) {
      int idx = tid + i * 256;
      int kr  = idx >> 4;           // 0..31
      int n   = (idx & 15) * 8;     // 0..120
      breg[i] = *(const v8h*)(Bm + (size_t)(k0 + kr) * N + col0 + n);
    }
  };
  auto store_B = [&](int buf) {
    #pragma unroll
    for (int i = 0; i < 2; ++i) {
      int idx = tid + i * 256;
      int kr  = idx >> 4;
      int n   = (idx & 15) * 8;
      #pragma unroll
      for (int j = 0; j < 8; ++j) Bt[buf][n + j][kr] = breg[i][j];
    }
  };

  v8f acc[2][4];
  #pragma unroll
  for (int i = 0; i < 2; ++i)
    #pragma unroll
    for (int j = 0; j < 4; ++j)
      acc[i][j] = 0.0f;

  // ---- prologue: fill buffer 0 ----
  stage_A(0, 0);
  load_B(0);
  store_B(0);
  wait_A();
  __syncthreads();

  const int nk = K >> 5;
  for (int t = 0; t < nk; ++t) {
    const int  cur  = t & 1;
    const int  nxt  = cur ^ 1;
    const bool more = (t + 1 < nk);

    if (more) {                    // kick off next tile before computing
      stage_A(nxt, (t + 1) * 32);  // TDM DMA (async) or direct copy
      load_B((t + 1) * 32);        // global loads issued now, used later
    }

    // ---- fragments (ISA 7.12.2 layouts) + 8 WMMAs on current buffers ----
    v16h af[2], bf[4];
    #pragma unroll
    for (int sm = 0; sm < 2; ++sm) {
      int r = wm * 32 + sm * 16 + l16;
      af[sm] = cat8(*(const v8h*)&As[cur][r][hf * 8],
                    *(const v8h*)&As[cur][r][16 + hf * 8]);
    }
    #pragma unroll
    for (int sn = 0; sn < 4; ++sn) {
      int c = wn * 64 + sn * 16 + l16;
      bf[sn] = cat8(*(const v8h*)&Bt[cur][c][hf * 8],
                    *(const v8h*)&Bt[cur][c][16 + hf * 8]);
    }
    #pragma unroll
    for (int sm = 0; sm < 2; ++sm)
      #pragma unroll
      for (int sn = 0; sn < 4; ++sn)
        acc[sm][sn] = wmma_f16(af[sm], bf[sn], acc[sm][sn]);

    if (more) {
      store_B(nxt);                // B regs -> LDS after compute
      wait_A();                    // TDM done before releasing barrier
    }
    __syncthreads();
  }

  // ---- epilogue: C layout lane n = l16, VGPR r -> row r + 8*hf ----
  #pragma unroll
  for (int sm = 0; sm < 2; ++sm) {
    #pragma unroll
    for (int sn = 0; sn < 4; ++sn) {
      int gm_base = row0 + wm * 32 + sm * 16;
      int gn      = col0 + wn * 64 + sn * 16 + l16;
      float bv = bias ? bias[gn] : 0.0f;
      #pragma unroll
      for (int r = 0; r < 8; ++r) {
        int gm = gm_base + r + hf * 8;
        float v = (acc[sm][sn][r] + bv) * scale;
        if (GELU) v = gelu_tanh_f(v);
        if (OUT_HALF) ((h16*)outp)[(size_t)gm * N + gn] = (h16)v;
        else          ((float*)outp)[(size_t)gm * N + gn] = v;
      }
    }
  }
}

// ---------------------------------------------------------------------------
// Flash attention: grid (S/64, NH, B), 128 threads = 4 waves.
// Each wave owns 16 queries; key loop in 64-key tiles, online softmax.
// q is pre-scaled by 1/sqrt(HD) in the QKV GEMM epilogue.
// ---------------------------------------------------------------------------
__global__ __launch_bounds__(128)
void flash_attn(const h16* __restrict__ q, const h16* __restrict__ k,
                const h16* __restrict__ v, const float* __restrict__ mask,
                h16* __restrict__ ctx)
{
  __shared__ h16 Vt[64][72];       // V tile transposed: [d][key]
  __shared__ h16 Pl[4][16][72];    // per-wave P tile: [m][key]

  const int tid  = threadIdx.x;
  const int lane = tid & 31;
  const int wid  = tid >> 5;
  const int l16  = lane & 15;
  const int hf   = lane >> 4;
  const int qt = blockIdx.x, hh = blockIdx.y, bb = blockIdx.z;
  const size_t baseQ = (size_t)bb * S_ * H_ + (size_t)hh * HD_;

  // Q fragments (held for the whole kernel)
  const int qrow = qt * 64 + wid * 16 + l16;
  v16h qf[2];
  {
    const h16* qp = q + baseQ + (size_t)qrow * H_;
    #pragma unroll
    for (int kc = 0; kc < 2; ++kc) {
      v8h lo = *(const v8h*)(qp + kc * 32 + hf * 8);
      v8h hi = *(const v8h*)(qp + kc * 32 + 16 + hf * 8);
      qf[kc] = cat8(lo, hi);
    }
  }

  v8f o[4];
  #pragma unroll
  for (int t = 0; t < 4; ++t) o[t] = 0.0f;
  float mrow[8], lrow[8];
  #pragma unroll
  for (int r = 0; r < 8; ++r) { mrow[r] = -3.0e38f; lrow[r] = 0.0f; }

  for (int j = 0; j < S_ / 64; ++j) {
    // prefetch next K/V tiles (one row per lane covers 64 rows)
    if (j + 1 < S_ / 64) {
      size_t nb = baseQ + (size_t)((j + 1) * 64) * H_;
      __builtin_prefetch(k + nb + (size_t)lane * H_, 0, 1);
      __builtin_prefetch(k + nb + (size_t)(32 + lane) * H_, 0, 1);
      __builtin_prefetch(v + nb + (size_t)lane * H_, 0, 1);
      __builtin_prefetch(v + nb + (size_t)(32 + lane) * H_, 0, 1);
    }

    // ---- cooperative V-tile transpose into LDS ----
    __syncthreads();
    #pragma unroll
    for (int i = 0; i < 4; ++i) {
      int idx = tid + i * 128;         // 512 chunks of 8 halves
      int key = idx >> 3;              // 0..63
      int c   = (idx & 7) * 8;         // 0..56
      v8h vv = *(const v8h*)(v + baseQ + (size_t)(j * 64 + key) * H_ + c);
      #pragma unroll
      for (int jj = 0; jj < 8; ++jj) Vt[c + jj][key] = vv[jj];
    }
    __syncthreads();

    // ---- S = Q @ K^T over this 64-key tile (4 tiles of 16 keys) ----
    v8f s[4];
    #pragma unroll
    for (int sn = 0; sn < 4; ++sn) {
      int key = j * 64 + sn * 16 + l16;
      const h16* kp = k + baseQ + (size_t)key * H_;
      v16h kf0 = cat8(*(const v8h*)(kp + hf * 8),      *(const v8h*)(kp + 16 + hf * 8));
      v16h kf1 = cat8(*(const v8h*)(kp + 32 + hf * 8), *(const v8h*)(kp + 32 + 16 + hf * 8));
      v8f z = 0.0f;
      z = wmma_f16(qf[0], kf0, z);
      z = wmma_f16(qf[1], kf1, z);
      float mv = mask[(size_t)bb * S_ + key];
      #pragma unroll
      for (int r = 0; r < 8; ++r) z[r] += mv;
      s[sn] = z;
    }

    // ---- online softmax: half-wave row reductions (rows m = r + 8*hf) ----
    float mnew[8], alpha[8];
    #pragma unroll
    for (int r = 0; r < 8; ++r) {
      float mx = fmaxf(fmaxf(s[0][r], s[1][r]), fmaxf(s[2][r], s[3][r]));
      #pragma unroll
      for (int off = 8; off >= 1; off >>= 1) mx = fmaxf(mx, __shfl_xor(mx, off, 32));
      mnew[r]  = fmaxf(mrow[r], mx);
      alpha[r] = __expf(mrow[r] - mnew[r]);
      mrow[r]  = mnew[r];
    }
    #pragma unroll
    for (int sn = 0; sn < 4; ++sn)
      #pragma unroll
      for (int r = 0; r < 8; ++r) {
        float p = __expf(s[sn][r] - mnew[r]);
        s[sn][r] = p;
        Pl[wid][r + hf * 8][sn * 16 + l16] = (h16)p;
      }
    #pragma unroll
    for (int r = 0; r < 8; ++r) {
      float rs = s[0][r] + s[1][r] + s[2][r] + s[3][r];
      #pragma unroll
      for (int off = 8; off >= 1; off >>= 1) rs += __shfl_xor(rs, off, 32);
      lrow[r] = lrow[r] * alpha[r] + rs;
    }
    #pragma unroll
    for (int t = 0; t < 4; ++t)
      #pragma unroll
      for (int r = 0; r < 8; ++r) o[t][r] *= alpha[r];

    // same-wave LDS RAW: be explicit about the DS counter
    asm volatile("s_wait_dscnt 0" ::: "memory");

    // ---- P fragments (A layout) from per-wave LDS ----
    v16h pf[2];
    #pragma unroll
    for (int kc = 0; kc < 2; ++kc) {
      const h16* pp = &Pl[wid][l16][0];
      v8h lo = *(const v8h*)(pp + kc * 32 + hf * 8);
      v8h hi = *(const v8h*)(pp + kc * 32 + 16 + hf * 8);
      pf[kc] = cat8(lo, hi);
    }
    // ---- O += P @ V ----
    #pragma unroll
    for (int t = 0; t < 4; ++t) {
      const h16* vp = &Vt[t * 16 + l16][0];
      v16h vf0 = cat8(*(const v8h*)(vp + hf * 8),      *(const v8h*)(vp + 16 + hf * 8));
      v16h vf1 = cat8(*(const v8h*)(vp + 32 + hf * 8), *(const v8h*)(vp + 32 + 16 + hf * 8));
      o[t] = wmma_f16(pf[0], vf0, o[t]);
      o[t] = wmma_f16(pf[1], vf1, o[t]);
    }
  }

  // ---- normalize + write ctx (f16, [b,s,h*64+d] layout) ----
  #pragma unroll
  for (int t = 0; t < 4; ++t)
    #pragma unroll
    for (int r = 0; r < 8; ++r) {
      int m = qt * 64 + wid * 16 + r + hf * 8;
      float val = o[t][r] / lrow[r];
      ctx[baseQ + (size_t)m * H_ + t * 16 + l16] = (h16)val;
    }
}

// ---------------------------------------------------------------------------
// Fused residual + LayerNorm: out = LN(X + Y) * g + b.  One wave per row.
// Optionally also writes an f16 copy (input of next GEMM).
// ---------------------------------------------------------------------------
template<bool WRITE_HALF>
__global__ __launch_bounds__(256)
void add_ln(const float* __restrict__ X, const float* __restrict__ Y,
            const float* __restrict__ g, const float* __restrict__ bt,
            float* __restrict__ outf, h16* __restrict__ outh)
{
  const int row  = blockIdx.x * 8 + (threadIdx.x >> 5);
  const int lane = threadIdx.x & 31;
  const float* xp = X + (size_t)row * H_;
  const float* yp = Y + (size_t)row * H_;

  float4 vals[8];
  float s = 0.0f, s2 = 0.0f;
  #pragma unroll
  for (int i = 0; i < 8; ++i) {
    int c = (lane + i * 32) * 4;
    float4 a = *(const float4*)(xp + c);
    float4 b = *(const float4*)(yp + c);
    float4 r;
    r.x = a.x + b.x; r.y = a.y + b.y; r.z = a.z + b.z; r.w = a.w + b.w;
    vals[i] = r;
    s  += r.x + r.y + r.z + r.w;
    s2 += r.x * r.x + r.y * r.y + r.z * r.z + r.w * r.w;
  }
  #pragma unroll
  for (int off = 16; off >= 1; off >>= 1) {
    s  += __shfl_xor(s,  off, 32);
    s2 += __shfl_xor(s2, off, 32);
  }
  const float inv = 1.0f / (float)H_;
  float mu   = s * inv;
  float var  = s2 * inv - mu * mu;
  float rstd = rsqrtf(var + 1e-12f);

  #pragma unroll
  for (int i = 0; i < 8; ++i) {
    int c = (lane + i * 32) * 4;
    float4 gg = *(const float4*)(g + c);
    float4 bb = *(const float4*)(bt + c);
    float o0 = (vals[i].x - mu) * rstd * gg.x + bb.x;
    float o1 = (vals[i].y - mu) * rstd * gg.y + bb.y;
    float o2 = (vals[i].z - mu) * rstd * gg.z + bb.z;
    float o3 = (vals[i].w - mu) * rstd * gg.w + bb.w;
    float4 ov; ov.x = o0; ov.y = o1; ov.z = o2; ov.w = o3;
    *(float4*)(outf + (size_t)row * H_ + c) = ov;
    if (WRITE_HALF) {
      v4h hh; hh[0] = (h16)o0; hh[1] = (h16)o1; hh[2] = (h16)o2; hh[3] = (h16)o3;
      *(v4h*)(outh + (size_t)row * H_ + c) = hh;
    }
  }
}

// ---------------------------------------------------------------------------
// Host orchestration
// ---------------------------------------------------------------------------
extern "C" void kernel_launch(void* const* d_in, const int* in_sizes, int n_in,
                              void* d_out, int out_size, void* d_ws, size_t ws_size,
                              hipStream_t stream) {
  (void)in_sizes; (void)n_in; (void)out_size; (void)ws_size;

  const float* x     = (const float*)d_in[0];
  const float* mask  = (const float*)d_in[1];
  const float* Wq    = (const float*)d_in[2];
  const float* bq    = (const float*)d_in[3];
  const float* Wk    = (const float*)d_in[4];
  const float* bk    = (const float*)d_in[5];
  const float* Wv    = (const float*)d_in[6];
  const float* bv    = (const float*)d_in[7];
  const float* Wo    = (const float*)d_in[8];
  const float* bo    = (const float*)d_in[9];
  const float* ln1_g = (const float*)d_in[10];
  const float* ln1_b = (const float*)d_in[11];
  const float* W1    = (const float*)d_in[12];
  const float* b1    = (const float*)d_in[13];
  const float* W2    = (const float*)d_in[14];
  const float* b2    = (const float*)d_in[15];
  const float* ln2_g = (const float*)d_in[16];
  const float* ln2_b = (const float*)d_in[17];

  char* ws = (char*)d_ws;
  const size_t MB = (size_t)1 << 20;
  // bump layout with reuse (peak ~168 MB)
  h16*   xh   = (h16*)(ws + 0 * MB);     // 16 MB, dead after QKV GEMMs
  h16*   wqh  = (h16*)(ws + 16 * MB);    // 2 MB
  h16*   wkh  = (h16*)(ws + 18 * MB);
  h16*   wvh  = (h16*)(ws + 20 * MB);
  h16*   woh  = (h16*)(ws + 22 * MB);
  h16*   w1h  = (h16*)(ws + 24 * MB);    // 8 MB
  h16*   w2h  = (h16*)(ws + 32 * MB);    // 8 MB
  h16*   qh   = (h16*)(ws + 40 * MB);    // 16 MB each; q/k/v/ctx dead after Wo GEMM
  h16*   kh   = (h16*)(ws + 56 * MB);
  h16*   vh   = (h16*)(ws + 72 * MB);
  h16*   ctxh = (h16*)(ws + 88 * MB);
  float* attn = (float*)(ws + 104 * MB); // 32 MB, dead after LN1
  float* x1f  = (float*)(ws + 136 * MB); // 32 MB
  h16*   x1h  = xh;                      // reuse (xh dead)
  h16*   ffh  = qh;                      // 64 MB over q/k/v/ctx region (dead)
  float* ff2  = attn;                    // reuse (attn dead)
  float* outp = (float*)d_out;

  auto cvt = [&](const float* in, h16* out, int n) {
    int n4 = n / 4;
    cvt4_kernel<<<(n4 + 255) / 256, 256, 0, stream>>>((const float4*)in, out, n4);
  };
  cvt(x,  xh,  M_ * H_);
  cvt(Wq, wqh, H_ * H_);
  cvt(Wk, wkh, H_ * H_);
  cvt(Wv, wvh, H_ * H_);
  cvt(Wo, woh, H_ * H_);
  cvt(W1, w1h, H_ * FF_);
  cvt(W2, w2h, FF_ * H_);

  dim3 blk(256);
  dim3 gHH(H_ / 128, M_ / 128);
  dim3 gFF(FF_ / 128, M_ / 128);

  // QKV projections (q pre-scaled by 1/sqrt(HD)=0.125)
  gemm_f16<true, false><<<gHH, blk, 0, stream>>>(xh, wqh, bq, qh, M_, H_, H_, 0.125f);
  gemm_f16<true, false><<<gHH, blk, 0, stream>>>(xh, wkh, bk, kh, M_, H_, H_, 1.0f);
  gemm_f16<true, false><<<gHH, blk, 0, stream>>>(xh, wvh, bv, vh, M_, H_, H_, 1.0f);

  // flash attention -> ctx (f16)
  flash_attn<<<dim3(S_ / 64, NH_, B_), dim3(128), 0, stream>>>(qh, kh, vh, mask, ctxh);

  // output projection -> attn (f32)
  gemm_f16<false, false><<<gHH, blk, 0, stream>>>(ctxh, woh, bo, attn, M_, H_, H_, 1.0f);

  // x1 = LN(x + attn)  (f32 + f16 copies)
  add_ln<true><<<M_ / 8, blk, 0, stream>>>(x, attn, ln1_g, ln1_b, x1f, x1h);

  // FFN: gelu(x1 @ W1 + b1) @ W2 + b2
  gemm_f16<true, true ><<<gFF, blk, 0, stream>>>(x1h, w1h, b1, ffh, M_, FF_, H_, 1.0f);
  gemm_f16<false, false><<<gHH, blk, 0, stream>>>(ffh, w2h, b2, ff2, M_, H_, FF_, 1.0f);

  // out = LN(x1 + ff)
  add_ln<false><<<M_ / 8, blk, 0, stream>>>(x1f, ff2, ln2_g, ln2_b, outp, nullptr);
}
